// ScalableGATLayer_27015344292632
// MI455X (gfx1250) — compile-verified, matrix operand order checked by reading, not run on hardware
//
#include <hip/hip_runtime.h>
#include <stddef.h>

// ---- problem constants ----
#define N_NODES   50000
#define N_EDGES   800000
#define ETOT      (N_EDGES + N_NODES)   // self loops appended
#define CH        128                    // in = out channels
#define HEADS     4
#define HDIM      32
#define MTILES    (N_NODES / 16)         // 3125 (exact)
#define NTILES    (CH / 16)              // 8
#define KSTEPS    (CH / 32)              // 4
#define NEG_SLOPE 0.2f
#define BN_EPS    1e-5f

typedef __attribute__((ext_vector_type(16))) _Float16 v16h;
typedef __attribute__((ext_vector_type(8)))  float    v8f;
typedef __attribute__((ext_vector_type(4)))  float    v4f;

// ---------- helpers ----------
__device__ __forceinline__ void atomAddF(float* p, float v) {
    __hip_atomic_fetch_add(p, v, __ATOMIC_RELAXED, __HIP_MEMORY_SCOPE_AGENT);
}
// order-preserving float -> uint encoding for atomic max
__device__ __forceinline__ unsigned fOrd(float f) {
    unsigned u = __float_as_uint(f);
    return (u & 0x80000000u) ? ~u : (u | 0x80000000u);
}
__device__ __forceinline__ float ordToF(unsigned u) {
    unsigned v = (u & 0x80000000u) ? (u & 0x7FFFFFFFu) : ~u;
    return __uint_as_float(v);
}

// ---------- 0: zero fill ----------
__global__ void zero_f32(float* __restrict__ p, int n) {
    int i = blockIdx.x * blockDim.x + threadIdx.x;
    if (i < n) p[i] = 0.0f;
}

// ---------- 1: pre-swizzle W (128x128 f32, row-major K x N) into WMMA B fragments ----
// B fragment layout (16-bit, 32x16, wave32): lane l holds col n=l&15; koff=(l>>4)*8;
// elements 0..7 -> K=koff+i, elements 8..15 -> K=16+koff+(i-8)   (ISA 7.12.2)
__global__ void prep_bfrag(const float* __restrict__ Wl, const float* __restrict__ Wr,
                           _Float16* __restrict__ fragL, _Float16* __restrict__ fragR) {
    int t = blockIdx.x * blockDim.x + threadIdx.x;      // 0..2047
    if (t >= 2 * NTILES * KSTEPS * 32) return;
    int mat  = t >> 10;
    int r    = t & 1023;                                 // (ntile*KSTEPS+ks)*32 + lane
    int lane = r & 31;
    int nk   = r >> 5;
    int ks   = nk & (KSTEPS - 1);
    int ntile = nk >> 2;
    const float* W   = mat ? Wr : Wl;
    _Float16*   frag = mat ? fragR : fragL;
    int n     = ntile * 16 + (lane & 15);
    int koff  = (lane >> 4) * 8;
    int kbase = ks * 32;
#pragma unroll
    for (int i = 0; i < 16; ++i) {
        int k = kbase + ((i < 8) ? (koff + i) : (16 + koff + (i - 8)));
        frag[(size_t)r * 16 + i] = (_Float16)W[(size_t)k * CH + n];
    }
}

// ---------- 2: GEMM  y = x @ W + b  via V_WMMA_F32_16X16X32_F16, f16 output ----------
// grid.x = MTILES, grid.y = 2 (lin_l / lin_r), 256 threads = 8 waves, one 16x16 tile/wave
__global__ void gemm_wmma(const float* __restrict__ x,
                          const _Float16* __restrict__ fragL, const _Float16* __restrict__ fragR,
                          const float* __restrict__ bl, const float* __restrict__ br,
                          _Float16* __restrict__ yl, _Float16* __restrict__ yr) {
    const _Float16* frag = blockIdx.y ? fragR : fragL;
    const float*    bvec = blockIdx.y ? br    : bl;
    _Float16*       y16  = blockIdx.y ? yr    : yl;

    int lane  = threadIdx.x & 31;
    int ntile = threadIdx.x >> 5;            // 0..7
    int mtile = blockIdx.x;                  // 0..3124
    int m     = mtile * 16 + (lane & 15);    // A row for this lane
    int koffA = (lane >> 4) * 8;             // A-frag K sub-offset (ISA 7.12.2)

    v8f acc = {0.f,0.f,0.f,0.f,0.f,0.f,0.f,0.f};
#pragma unroll
    for (int ks = 0; ks < KSTEPS; ++ks) {
        // A fragment: elements 0..7 = K koff..koff+7, 8..15 = K 16+koff..23+koff
        const float* ap = x + (size_t)m * CH + ks * 32 + koffA;
        float af[16];
        *(v4f*)(af + 0)  = *(const v4f*)(ap + 0);
        *(v4f*)(af + 4)  = *(const v4f*)(ap + 4);
        *(v4f*)(af + 8)  = *(const v4f*)(ap + 16);
        *(v4f*)(af + 12) = *(const v4f*)(ap + 20);
        v16h A;
#pragma unroll
        for (int i = 0; i < 16; ++i) A[i] = (_Float16)af[i];
        // B fragment: coalesced 32B load from pre-swizzled buffer
        v16h B = *(const v16h*)(frag + ((size_t)((ntile * KSTEPS + ks) * 32 + lane) << 4));
        acc = __builtin_amdgcn_wmma_f32_16x16x32_f16(false, A, false, B,
                                                     (short)0, acc, false, false);
    }
    // D layout: VGPR j -> row j (+8 for lanes 16..31), col = lane&15
    int col   = ntile * 16 + (lane & 15);
    int rbase = mtile * 16 + ((lane >> 4) << 3);
    float bv  = bvec[col];
#pragma unroll
    for (int j = 0; j < 8; ++j)
        y16[(size_t)(rbase + j) * CH + col] = (_Float16)(acc[j] + bv);
}

// ---------- 3: per-edge GATv2 scores + segment max (one wave32 per edge) ----------
__global__ void score_kernel(const int* __restrict__ ei,
                             const _Float16* __restrict__ xl16, const _Float16* __restrict__ xr16,
                             const float* __restrict__ att,
                             float* __restrict__ score, unsigned* __restrict__ m_u) {
    int wid  = (blockIdx.x * blockDim.x + threadIdx.x) >> 5;
    int lane = threadIdx.x & 31;
    if (wid >= ETOT) return;
    int src, dst;
    if (wid < N_EDGES) { src = ei[wid]; dst = ei[N_EDGES + wid]; }
    else               { src = dst = wid - N_EDGES; }

    const _Float16* pl = xl16 + (size_t)src * CH;
    const _Float16* pr = xr16 + (size_t)dst * CH;
    float s[HEADS];
#pragma unroll
    for (int h = 0; h < HEADS; ++h) {
        int c = h * HDIM + lane;
        float v = (float)pl[c] + (float)pr[c];
        v = v > 0.f ? v : NEG_SLOPE * v;
        s[h] = v * att[c];                       // att is (HEADS, HDIM) row-major
    }
#pragma unroll
    for (int off = 16; off > 0; off >>= 1) {
#pragma unroll
        for (int h = 0; h < HEADS; ++h) s[h] += __shfl_xor(s[h], off, 32);
    }
    if (lane == 0) {
#pragma unroll
        for (int h = 0; h < HEADS; ++h) {
            score[(size_t)wid * HEADS + h] = s[h];
            atomicMax(&m_u[dst * HEADS + h], fOrd(s[h]));
        }
    }
}

// ---------- 4: exp(score - max) and segment sum; overwrite score with exp value ----------
__global__ void expsum_kernel(const int* __restrict__ ei, float* __restrict__ score,
                              const unsigned* __restrict__ m_u, float* __restrict__ denom) {
    int idx = blockIdx.x * blockDim.x + threadIdx.x;
    if (idx >= ETOT * HEADS) return;
    int e = idx >> 2, h = idx & 3;
    int dst = (e < N_EDGES) ? ei[N_EDGES + e] : (e - N_EDGES);
    float m  = ordToF(m_u[dst * HEADS + h]);
    float ex = __expf(score[idx] - m);
    score[idx] = ex;
    atomAddF(&denom[dst * HEADS + h], ex);
}

// ---------- 5: scatter-add alpha * x_l[src] into out (one wave32 per edge) ----------
__global__ void agg_kernel(const int* __restrict__ ei, const float* __restrict__ score,
                           const float* __restrict__ denom, const _Float16* __restrict__ xl16,
                           float* __restrict__ out) {
    int wid  = (blockIdx.x * blockDim.x + threadIdx.x) >> 5;
    int lane = threadIdx.x & 31;
    if (wid >= ETOT) return;
    int src, dst;
    if (wid < N_EDGES) { src = ei[wid]; dst = ei[N_EDGES + wid]; }
    else               { src = dst = wid - N_EDGES; }

    const _Float16* pl = xl16 + (size_t)src * CH;
    float* po = out + (size_t)dst * CH;
#pragma unroll
    for (int h = 0; h < HEADS; ++h) {
        float alpha = score[(size_t)wid * HEADS + h] / denom[dst * HEADS + h];
        int c = h * HDIM + lane;
        atomAddF(po + c, alpha * (float)pl[c]);
    }
}

// ---------- 6: BatchNorm statistics (per-channel sum & sumsq) ----------
__global__ void bnstats_kernel(const float* __restrict__ out,
                               float* __restrict__ sums, float* __restrict__ sumsq) {
    int c = threadIdx.x;                     // blockDim.x == 128
    float s1 = 0.f, s2 = 0.f;
    for (int r = blockIdx.x; r < N_NODES; r += gridDim.x) {
        float v = out[(size_t)r * CH + c];
        s1 += v; s2 += v * v;
    }
    atomAddF(&sums[c], s1);
    atomAddF(&sumsq[c], s2);
}

// ---------- 7: BN normalize + residual + ELU (bias cancels inside BN) ----------
__global__ void final_kernel(float* __restrict__ out, const float* __restrict__ x,
                             const float* __restrict__ sums, const float* __restrict__ sumsq,
                             const float* __restrict__ gamma, const float* __restrict__ beta) {
    int idx = blockIdx.x * blockDim.x + threadIdx.x;
    if (idx >= N_NODES * CH) return;
    int c = idx & (CH - 1);
    const float invN = 1.0f / (float)N_NODES;
    float mu  = sums[c] * invN;
    float var = sumsq[c] * invN - mu * mu;
    float g   = gamma[c] * rsqrtf(var + BN_EPS);
    float y   = (out[idx] - mu) * g + beta[c] + x[idx];
    out[idx]  = y > 0.f ? y : expm1f(y);
}

// ---------- launch ----------
extern "C" void kernel_launch(void* const* d_in, const int* in_sizes, int n_in,
                              void* d_out, int out_size, void* d_ws, size_t ws_size,
                              hipStream_t stream) {
    const float* x    = (const float*)d_in[0];
    const int*   ei   = (const int*)  d_in[1];
    const float* Wl   = (const float*)d_in[2];
    const float* bl   = (const float*)d_in[3];
    const float* Wr   = (const float*)d_in[4];
    const float* br   = (const float*)d_in[5];
    const float* att  = (const float*)d_in[6];
    // d_in[7] = bias: cancels inside BatchNorm (per-channel constant)
    const float* gamma = (const float*)d_in[8];
    const float* beta  = (const float*)d_in[9];
    float* out = (float*)d_out;

    // workspace carve-up (all chunks 256B-multiples)
    char* w = (char*)d_ws;
    _Float16* fragL = (_Float16*)w;  w += (size_t)NTILES * KSTEPS * 32 * 16 * 2;  // 32 KB
    _Float16* fragR = (_Float16*)w;  w += (size_t)NTILES * KSTEPS * 32 * 16 * 2;  // 32 KB
    _Float16* xl16  = (_Float16*)w;  w += (size_t)N_NODES * CH * 2;               // 12.8 MB
    _Float16* xr16  = (_Float16*)w;  w += (size_t)N_NODES * CH * 2;               // 12.8 MB
    float*    score = (float*)w;     w += (size_t)ETOT * HEADS * 4;               // 13.6 MB
    unsigned* m_u   = (unsigned*)w;  w += (size_t)N_NODES * HEADS * 4;            // contiguous ↓
    float*    denom = (float*)w;     w += (size_t)N_NODES * HEADS * 4;
    float*    sums  = (float*)w;     w += 512;
    float*    sumsq = (float*)w;     w += 512;

    // zero accumulators: d_out (scatter target) and [m_u|denom|sums|sumsq] (contiguous)
    {
        int n0 = N_NODES * CH;
        zero_f32<<<(n0 + 255) / 256, 256, 0, stream>>>(out, n0);
        int n1 = N_NODES * HEADS * 2 + 256;  // m_u + denom + sums + sumsq (fp32-sized)
        zero_f32<<<(n1 + 255) / 256, 256, 0, stream>>>((float*)m_u, n1);
    }

    prep_bfrag<<<8, 256, 0, stream>>>(Wl, Wr, fragL, fragR);

    gemm_wmma<<<dim3(MTILES, 2), 256, 0, stream>>>(x, fragL, fragR, bl, br, xl16, xr16);

    {
        int blocks = (ETOT * 32 + 255) / 256;   // one wave per edge
        score_kernel<<<blocks, 256, 0, stream>>>(ei, xl16, xr16, att, score, m_u);
    }
    {
        int n = ETOT * HEADS;
        expsum_kernel<<<(n + 255) / 256, 256, 0, stream>>>(ei, score, m_u, denom);
    }
    {
        int blocks = (ETOT * 32 + 255) / 256;
        agg_kernel<<<blocks, 256, 0, stream>>>(ei, score, denom, xl16, out);
    }
    bnstats_kernel<<<512, 128, 0, stream>>>(out, sums, sumsq);
    {
        int n = N_NODES * CH;
        final_kernel<<<(n + 255) / 256, 256, 0, stream>>>(out, x, sums, sumsq, gamma, beta);
    }
}
